// RtoVMainModel_64287070486710
// MI455X (gfx1250) — matrix operand order repaired and checked by reading, not compiled, not run on hardware
//
#include <hip/hip_runtime.h>

// ---------------------------------------------------------------------------
// CDNA5 (gfx1250) wave32 WMMA implementation of the LeNet+experts pipeline.
//   k_conv  : conv1+relu+pool + conv2+relu+pool fully fused in LDS (one
//             image per block), im2col GEMM tiles on v_wmma_f32_16x16x32_bf16.
//   k_mlp   : one wave / 16-sample tile through shared head + 4 expert
//             chains, 128-bit LDS/global fragment loads, dual accumulators.
//   routing : batch-global "last covering sample" via atomicMax + finalize.
// ---------------------------------------------------------------------------

typedef __attribute__((ext_vector_type(16))) __bf16          v16bf;
typedef __attribute__((ext_vector_type(16))) unsigned short  v16u;
typedef __attribute__((ext_vector_type(8)))  float           v8f;

union Frag { v16u u; v16bf b; uint4 q[2]; };

__device__ __forceinline__ unsigned short bfbits(float f) {
  union { float f; unsigned u; } v; v.f = f;
  unsigned r = v.u + 0x7FFFu + ((v.u >> 16) & 1u);   // round-to-nearest-even
  return (unsigned short)(r >> 16);
}

__device__ __forceinline__ v8f wmma_bf16(const Frag& a, const Frag& b, v8f c) {
  return __builtin_amdgcn_wmma_f32_16x16x32_bf16(false, a.b, false, b.b,
                                                 (short)0, c, false, false);
}

// A-matrix (16x32 bf16) lane layout: lane L row m=L&15,
//   element e -> k = (L>>4)*16 + e
// => two contiguous 16B runs. Requires stride % 8 == 0, base 16B aligned.
__device__ __forceinline__ Frag loadA(const unsigned short* lds, int stride,
                                      int kc, int lane) {
  const int m = lane & 15, hi = lane >> 4;
  const unsigned short* p = lds + m * stride + kc * 32 + (hi << 4);
  Frag f;
  f.q[0] = *(const uint4*)p;
  f.q[1] = *(const uint4*)(p + 8);
  return f;
}

// B-matrix (32x16 bf16, B[k][n]=W[n][k]) from pre-converted zero-padded bf16
// weights Wb[Np][Kp]: lane L col n=L&15, element e -> k=(L>>4)*16+e.
__device__ __forceinline__ Frag loadBp(const unsigned short* __restrict__ Wb,
                                       int Kp, int n0, int k0, int lane) {
  const unsigned short* p = Wb + (size_t)(n0 + (lane & 15)) * Kp
                               + k0 + ((lane >> 4) << 4);
  Frag f;
  f.q[0] = *(const uint4*)p;
  f.q[1] = *(const uint4*)(p + 8);
  return f;
}

// FC layer over an even number of 16-wide N tiles, two tiles per iteration.
__device__ __forceinline__ void fc2(const unsigned short* in, int istride, int kch,
                                    const unsigned short* __restrict__ Wb, int Kp,
                                    const float* __restrict__ bias, int N, int ntiles,
                                    unsigned short* out, int ostride, int lane) {
  const int n16 = lane & 15, hi = lane >> 4;
  for (int nt = 0; nt < ntiles; nt += 2) {
    v8f acc0 = {}, acc1 = {};
    for (int kc = 0; kc < kch; ++kc) {
      Frag a  = loadA(in, istride, kc, lane);
      Frag b0 = loadBp(Wb, Kp, nt * 16,      kc * 32, lane);
      Frag b1 = loadBp(Wb, Kp, nt * 16 + 16, kc * 32, lane);
      acc0 = wmma_bf16(a, b0, acc0);
      acc1 = wmma_bf16(a, b1, acc1);
    }
    const int nA = nt * 16 + n16, nB = nA + 16;
    const float bvA = (nA < N) ? bias[nA] : 0.f;
    const float bvB = (nB < N) ? bias[nB] : 0.f;
#pragma unroll
    for (int r = 0; r < 8; ++r) {
      int m = r + hi * 8;
      out[m * ostride + nA] = bfbits((nA < N) ? (acc0[r] + bvA) : 0.f);
      out[m * ostride + nB] = bfbits((nB < N) ? (acc1[r] + bvB) : 0.f);
    }
  }
}

// ---------------------------------------------------------------------------
// Weight prep: fp32 W[N][K] -> bf16 Wb[Np][Kp], zero padded.
// ---------------------------------------------------------------------------
__global__ __launch_bounds__(256) void k_prepw(const float* __restrict__ W,
                                               int N, int K,
                                               unsigned short* __restrict__ Wb,
                                               int Kp, int total) {
  for (int i = blockIdx.x * 256 + threadIdx.x; i < total; i += gridDim.x * 256) {
    int n = i / Kp, k = i % Kp;
    Wb[i] = (n < N && k < K) ? bfbits(W[(size_t)n * K + k]) : (unsigned short)0;
  }
}

// ---------------------------------------------------------------------------
// Fused conv stack, one image per block (256 threads = 8 waves):
//   conv1 3->6 5x5 +relu (WMMA: 49 tiles, K=75->3 chunks) -> pool -> LDS
//   conv2 6->16 5x5 +relu (WMMA: 7 tiles, K=150->5 chunks) -> pool -> xf
// No HBM roundtrip for the intermediate activations.
// ---------------------------------------------------------------------------
__global__ __launch_bounds__(256) void k_conv(const float* __restrict__ x,
                                              const float* __restrict__ w1,
                                              const float* __restrict__ b1,
                                              const float* __restrict__ w2,
                                              const float* __restrict__ b2,
                                              unsigned short* __restrict__ xf) {
  __shared__ __align__(16) unsigned short img[4096];    // 3*32*32 + zero tail
  __shared__ float cbuf[6 * 28 * 28];                   // conv1 relu out
  __shared__ __align__(16) unsigned short img2[1408];   // pooled y1 + zero tail
  __shared__ float cbuf2[16 * 10 * 10];                 // conv2 relu out
  const int b = blockIdx.x;
  const int tid = threadIdx.x, lane = tid & 31, wid = tid >> 5;
  const int n = lane & 15, hi = lane >> 4;

  // ---- stage input image as bf16, vectorized (tail zeroed) ----
  const float* xb = x + (size_t)b * 3072;
  for (int i = tid * 4; i < 3072; i += 1024) {
    float4 v = *(const float4*)(xb + i);
    ushort4 o;
    o.x = bfbits(v.x); o.y = bfbits(v.y); o.z = bfbits(v.z); o.w = bfbits(v.w);
    *(ushort4*)(img + i) = o;
  }
  for (int i = 3072 + tid * 4; i < 4096; i += 1024) {
    ushort4 z = {};
    *(ushort4*)(img + i) = z;
  }
  __syncthreads();

  // ================= conv1 =================
  {
    Frag bw[3];
#pragma unroll
    for (int kc = 0; kc < 3; ++kc)
#pragma unroll
      for (int e = 0; e < 16; ++e) {
        int k = kc * 32 + hi * 16 + e;
        bw[kc].u[e] = (n < 6 && k < 75) ? bfbits(w1[n * 75 + k]) : (unsigned short)0;
      }
    const float bv = (n < 6) ? b1[n] : 0.f;

    int aoff[3][16];
#pragma unroll
    for (int kc = 0; kc < 3; ++kc)
#pragma unroll
      for (int e = 0; e < 16; ++e) {
        int k = kc * 32 + hi * 16 + e;
        int off = 3072;                                  // zero pad
        if (k < 75) {
          int ic = k / 25, rem = k % 25;
          off = ic * 1024 + (rem / 5) * 32 + (rem % 5);
        }
        aoff[kc][e] = off;
      }

    for (int t = wid; t < 49; t += 8) {
      const int p = t * 16 + (lane & 15);
      const int pbase = (p / 28) * 32 + (p % 28);        // oy*32 + ox
      Frag a[3];
#pragma unroll
      for (int kc = 0; kc < 3; ++kc)
#pragma unroll
        for (int e = 0; e < 16; ++e)
          a[kc].u[e] = img[pbase + aoff[kc][e]];
      v8f acc = {};
      acc = wmma_bf16(a[0], bw[0], acc);
      acc = wmma_bf16(a[1], bw[1], acc);
      acc = wmma_bf16(a[2], bw[2], acc);
      if (n < 6) {
#pragma unroll
        for (int r = 0; r < 8; ++r) {
          int pr = t * 16 + r + hi * 8;
          cbuf[n * 784 + pr] = fmaxf(acc[r] + bv, 0.f);
        }
      }
    }
  }
  __syncthreads();

  // ---- pool1 -> img2 (bf16, tail zeroed) ----
  for (int i = tid; i < 6 * 14 * 14; i += 256) {
    int oc = i / 196, rem = i % 196, py = rem / 14, px = rem % 14;
    const float* s = cbuf + oc * 784 + py * 2 * 28 + px * 2;
    float v = fmaxf(fmaxf(s[0], s[1]), fmaxf(s[28], s[29]));
    img2[i] = bfbits(v);
  }
  for (int i = tid + 1176; i < 1408; i += 256) img2[i] = 0;
  __syncthreads();

  // ================= conv2 =================
  {
    Frag bw[5];
#pragma unroll
    for (int kc = 0; kc < 5; ++kc)
#pragma unroll
      for (int e = 0; e < 16; ++e) {
        int k = kc * 32 + hi * 16 + e;
        bw[kc].u[e] = (k < 150) ? bfbits(w2[n * 150 + k]) : (unsigned short)0;
      }
    const float bv = b2[n];

    int aoff[5][16];
#pragma unroll
    for (int kc = 0; kc < 5; ++kc)
#pragma unroll
      for (int e = 0; e < 16; ++e) {
        int k = kc * 32 + hi * 16 + e;
        int off = 1176;                                  // zero pad
        if (k < 150) {
          int ic = k / 25, rem = k % 25;
          off = ic * 196 + (rem / 5) * 14 + (rem % 5);
        }
        aoff[kc][e] = off;
      }

    for (int t = wid; t < 7; t += 8) {
      const int p = t * 16 + (lane & 15);
      const bool pv = p < 100;
      const int pbase = pv ? ((p / 10) * 14 + (p % 10)) : 0;
      Frag a[5];
#pragma unroll
      for (int kc = 0; kc < 5; ++kc)
#pragma unroll
        for (int e = 0; e < 16; ++e)
          a[kc].u[e] = pv ? img2[pbase + aoff[kc][e]] : (unsigned short)0;
      v8f acc = {};
#pragma unroll
      for (int kc = 0; kc < 5; ++kc) acc = wmma_bf16(a[kc], bw[kc], acc);
#pragma unroll
      for (int r = 0; r < 8; ++r) {
        int pr = t * 16 + r + hi * 8;
        if (pr < 100) cbuf2[n * 100 + pr] = fmaxf(acc[r] + bv, 0.f);
      }
    }
  }
  __syncthreads();

  // ---- pool2 -> xf (B,400) bf16, layout oc*25+py*5+px ----
  for (int i = tid; i < 400; i += 256) {
    int oc = i / 25, rem = i % 25, py = rem / 5, px = rem % 5;
    const float* s = cbuf2 + oc * 100 + py * 2 * 10 + px * 2;
    float v = fmaxf(fmaxf(s[0], s[1]), fmaxf(s[10], s[11]));
    xf[(size_t)b * 400 + i] = bfbits(v);
  }
}

// ---------------------------------------------------------------------------
// MLP mega-kernel: one wave handles 16 samples through the shared head and
// all 4 expert chains. h1/h2 reused per chain (only h3 persists) -> ~25KB LDS.
// ---------------------------------------------------------------------------
__global__ __launch_bounds__(32) void k_mlp(const unsigned short* __restrict__ xf,
    const unsigned short* __restrict__ w1b, const unsigned short* __restrict__ w2b,
    const unsigned short* __restrict__ w3b,
    const float* __restrict__ sb1, const float* __restrict__ eb1,
    const float* __restrict__ sb2, const float* __restrict__ eb2,
    const float* __restrict__ sb3, const float* __restrict__ eb3,
    const float* __restrict__ sw4, const float* __restrict__ sb4,
    const float* __restrict__ ew4, const float* __restrict__ eb4,
    float* __restrict__ shapes_out, float* __restrict__ eo,
    int* __restrict__ pred, int* __restrict__ lastcov) {
  __shared__ __align__(16) unsigned short xfT[16 * 416];  // K 400 -> 416
  __shared__ __align__(16) unsigned short h1[16 * 128];   // N 120 -> 128 (reused)
  __shared__ __align__(16) unsigned short h2[16 * 96];    // N  84 -> 96  (reused)
  __shared__ float h3[5 * 16 * 16];                       // all chains (fp32)
  const int lane = threadIdx.x;
  const int s0 = blockIdx.x * 16;

  // ---- stage 16x400 tile as 8-element vectors (400 % 8 == 0, rows stay
  //      16B aligned in both global (800B) and padded LDS (832B) strides) ----
  {
    const unsigned short* xs = xf + (size_t)s0 * 400;
    for (int j = lane; j < 800; j += 32) {
      int m = j / 50, kb = (j % 50) * 8;
      *(uint4*)(xfT + m * 416 + kb) = *(const uint4*)(xs + m * 400 + kb);
    }
    uint4 z = {};
    const int m = lane >> 1, kb = 400 + (lane & 1) * 8;   // zero tail 400..415
    *(uint4*)(xfT + m * 416 + kb) = z;
  }
  __syncthreads();

  for (int src = 0; src < 5; ++src) {
    const float* B1 = (src == 0) ? sb1 : eb1 + (src - 1) * 120;
    fc2(xfT, 416, 13, w1b + src * 128 * 416, 416, B1, 120, 8, h1, 128, lane);

    const float* B2 = (src == 0) ? sb2 : eb2 + (src - 1) * 84;
    fc2(h1, 128, 4, w2b + src * 96 * 128, 128, B2, 84, 6, h2, 96, lane);

    const float* B3 = (src == 0) ? sb3 : eb3 + (src - 1) * 10;
    {
      v8f acc = {};
      for (int kc = 0; kc < 3; ++kc) {
        Frag a = loadA(h2, 96, kc, lane);
        Frag b = loadBp(w3b + src * 16 * 96, 96, 0, kc * 32, lane);
        acc = wmma_bf16(a, b, acc);
      }
      const int nn = lane & 15, hi = lane >> 4;
      const float bvv = (nn < 10) ? B3[nn] : 0.f;
#pragma unroll
      for (int r = 0; r < 8; ++r) {
        int m = r + hi * 8;
        h3[src * 256 + m * 16 + nn] = (nn < 10) ? (acc[r] + bvv) : 0.f;
      }
    }
  }
  __syncthreads();

  if (lane < 16) {
    const int m = lane, s = s0 + m;
    const float* hs = h3 + m * 16;          // shared head (chain 0)
    float sh[4];
#pragma unroll
    for (int j = 0; j < 4; ++j) {
      float acc = sb4[j];
#pragma unroll
      for (int i = 0; i < 10; ++i) acc += fmaxf(hs[i], 0.f) * sw4[j * 10 + i];
      sh[j] = acc;
      shapes_out[(size_t)s * 4 + j] = acc;
    }
    int p = 0; float best = sh[0];
#pragma unroll
    for (int j = 1; j < 4; ++j) if (sh[j] > best) { best = sh[j]; p = j; }
    pred[s] = p;
    const int widths[4] = {3, 5, 6, 5};
    const int wp = widths[p];
#pragma unroll
    for (int c = 0; c < 6; ++c) if (wp > c) atomicMax(&lastcov[c], s);
#pragma unroll
    for (int e = 0; e < 4; ++e) {
      const float* he = h3 + (1 + e) * 256 + m * 16;
#pragma unroll
      for (int c = 0; c < 6; ++c) {
        float acc = eb4[e * 6 + c];
#pragma unroll
        for (int i = 0; i < 10; ++i) acc += he[i] * ew4[(e * 6 + c) * 10 + i];
        eo[((size_t)e * 16384 + s) * 6 + c] = acc;
      }
    }
  }
}

__global__ void k_init(int* lastcov) {
  if (threadIdx.x < 6) lastcov[threadIdx.x] = -1;
}

__global__ __launch_bounds__(256) void k_final(const float* __restrict__ eo,
                                               const int* __restrict__ pred,
                                               const int* __restrict__ lastcov,
                                               float* __restrict__ out2) {
  const int b = blockIdx.x * 256 + threadIdx.x;
  if (b >= 16384) return;
#pragma unroll
  for (int c = 0; c < 6; ++c) {
    int lb = lastcov[c];
    float v = 0.f;
    if (lb >= 0) {
      int ej = pred[lb];
      v = eo[((size_t)ej * 16384 + b) * 6 + c];
    }
    out2[(size_t)b * 6 + c] = v;
  }
}

extern "C" void kernel_launch(void* const* d_in, const int* in_sizes, int n_in,
                              void* d_out, int out_size, void* d_ws, size_t ws_size,
                              hipStream_t stream) {
  (void)in_sizes; (void)n_in; (void)out_size; (void)ws_size;
  const float* x   = (const float*)d_in[0];
  const float* c1w = (const float*)d_in[1];
  const float* c1b = (const float*)d_in[2];
  const float* c2w = (const float*)d_in[3];
  const float* c2b = (const float*)d_in[4];
  const float* sw1 = (const float*)d_in[5];  const float* sb1 = (const float*)d_in[6];
  const float* sw2 = (const float*)d_in[7];  const float* sb2 = (const float*)d_in[8];
  const float* sw3 = (const float*)d_in[9];  const float* sb3 = (const float*)d_in[10];
  const float* sw4 = (const float*)d_in[11]; const float* sb4 = (const float*)d_in[12];
  const float* ew1 = (const float*)d_in[13]; const float* eb1 = (const float*)d_in[14];
  const float* ew2 = (const float*)d_in[15]; const float* eb2 = (const float*)d_in[16];
  const float* ew3 = (const float*)d_in[17]; const float* eb3 = (const float*)d_in[18];
  const float* ew4 = (const float*)d_in[19]; const float* eb4 = (const float*)d_in[20];

  // workspace layout (bytes, 256-aligned) -- ~15.4 MB total
  char* ws = (char*)d_ws;
  unsigned short* xf  = (unsigned short*)(ws);                 // 13,107,200
  float*          eo  = (float*)(ws + 13107200);               //  1,572,864
  int*            pr  = (int*)  (ws + 14680064);               //     65,536
  int*            lc  = (int*)  (ws + 14745600);               //        256
  unsigned short* w1b = (unsigned short*)(ws + 14745856);      // 5*128*416*2
  unsigned short* w2b = (unsigned short*)(ws + 15278336);      // 5* 96*128*2
  unsigned short* w3b = (unsigned short*)(ws + 15401216);      // 5* 16* 96*2

  float* shapes_out = (float*)d_out;
  float* out2       = (float*)d_out + (size_t)16384 * 4;

  k_init<<<1, 32, 0, stream>>>(lc);

  for (int s = 0; s < 5; ++s) {
    const float* W1 = (s == 0) ? sw1 : ew1 + (size_t)(s - 1) * 120 * 400;
    k_prepw<<<208, 256, 0, stream>>>(W1, 120, 400, w1b + s * 128 * 416, 416, 128 * 416);
    const float* W2 = (s == 0) ? sw2 : ew2 + (size_t)(s - 1) * 84 * 120;
    k_prepw<<<48, 256, 0, stream>>>(W2, 84, 120, w2b + s * 96 * 128, 128, 96 * 128);
    const float* W3 = (s == 0) ? sw3 : ew3 + (size_t)(s - 1) * 10 * 84;
    k_prepw<<<6, 256, 0, stream>>>(W3, 10, 84, w3b + s * 16 * 96, 96, 16 * 96);
  }

  k_conv<<<16384, 256, 0, stream>>>(x, c1w, c1b, c2w, c2b, xf);
  k_mlp <<<1024,  32, 0, stream>>>(xf, w1b, w2b, w3b,
                                   sb1, eb1, sb2, eb2, sb3, eb3,
                                   sw4, sb4, ew4, eb4,
                                   shapes_out, eo, pr, lc);
  k_final<<<64,   256, 0, stream>>>(eo, pr, lc, out2);
}
